// TransitionUp_35656818492087
// MI455X (gfx1250) — matrix operand order repaired and checked by reading, not compiled, not run on hardware
//
#include <hip/hip_runtime.h>
#include <hip/hip_bf16.h>

// ---------------------------------------------------------------------------
// Problem constants (from setup_inputs)
// ---------------------------------------------------------------------------
#define BATCH   4
#define N_LOW   4096
#define N_HIGH  16384
#define K_LOW   512
#define K_HIGH  256
#define D_OUT   256
#define BN_EPS  1e-5f

#define M_LOW   (BATCH * N_LOW)    // 16384
#define M_HIGH  (BATCH * N_HIGH)   // 65536

// Workspace layout (floats)
#define FL_ELEMS   ((size_t)M_LOW * D_OUT)          // 4,194,304 f32  (fl)
#define KNN_ELEMS  ((size_t)M_HIGH * 3)             // 196,608 (idx int / w float)

// ---------------------------------------------------------------------------
// CDNA5 WMMA types
// ---------------------------------------------------------------------------
typedef __attribute__((ext_vector_type(16))) __bf16 v16bf;
typedef __attribute__((ext_vector_type(8)))  float  v8f;

__device__ __forceinline__ __bf16 f2bf(float f) {
    union { float f; unsigned u; } v; v.f = f;
    unsigned r = v.u + 0x7FFFu + ((v.u >> 16) & 1u);   // round-to-nearest-even
    unsigned short h = (unsigned short)(r >> 16);
    __bf16 b;
    __builtin_memcpy(&b, &h, 2);
    return b;
}

// ---------------------------------------------------------------------------
// Fused GEMM + BatchNorm + ReLU (bf16 WMMA, f32 accumulate).
// Block tile 128(M) x 64(N) x 32(K), 8 waves in a 4x2 grid; each wave owns a
// 32x32 output patch = 2x2 WMMA tiles (fragment reuse: 4 wmma per 4 frags).
// Double-buffered LDS: global->reg prefetch overlaps WMMA; 1 barrier / K-step.
// ---------------------------------------------------------------------------
#define BM  128
#define BN  64
#define BK  32
#define LDA (BK + 8)   // +16B row pad: 16B frag reads hit distinct banks

__device__ __forceinline__ void gload(const float* __restrict__ X,
                                      const float* __restrict__ W,
                                      int m0, int n0, int K, int k0, int tid,
                                      float4* pa, float4* pb)
{
#pragma unroll
    for (int it = 0; it < 4; ++it) {            // A: 128 rows x 32 k = 1024 f4
        int c = tid + it * 256;
        pa[it] = *(const float4*)(X + (size_t)(m0 + (c >> 3)) * K + k0 + ((c & 7) << 2));
    }
#pragma unroll
    for (int it = 0; it < 2; ++it) {            // B: 64 rows x 32 k = 512 f4
        int c = tid + it * 256;
        pb[it] = *(const float4*)(W + (size_t)(n0 + (c >> 3)) * K + k0 + ((c & 7) << 2));
    }
}

__device__ __forceinline__ void lstore(__bf16 (*as)[LDA], __bf16 (*bs)[LDA],
                                       int tid, const float4* pa, const float4* pb)
{
#pragma unroll
    for (int it = 0; it < 4; ++it) {
        int c = tid + it * 256;
        int r = c >> 3, cc = (c & 7) << 2;
        as[r][cc + 0] = f2bf(pa[it].x);
        as[r][cc + 1] = f2bf(pa[it].y);
        as[r][cc + 2] = f2bf(pa[it].z);
        as[r][cc + 3] = f2bf(pa[it].w);
    }
#pragma unroll
    for (int it = 0; it < 2; ++it) {
        int c = tid + it * 256;
        int r = c >> 3, cc = (c & 7) << 2;
        bs[r][cc + 0] = f2bf(pb[it].x);
        bs[r][cc + 1] = f2bf(pb[it].y);
        bs[r][cc + 2] = f2bf(pb[it].z);
        bs[r][cc + 3] = f2bf(pb[it].w);
    }
}

__global__ __launch_bounds__(256)
void mlp_gemm_kernel(const float* __restrict__ X,      // [M][K]
                     const float* __restrict__ W,      // [D_OUT][K]
                     const float* __restrict__ bias,
                     const float* __restrict__ gamma,
                     const float* __restrict__ beta,
                     const float* __restrict__ mean,
                     const float* __restrict__ var,
                     float* __restrict__ Y,            // [M][D_OUT]
                     int M, int K)
{
    __shared__ __bf16 As[2][BM][LDA];   // 2 x 128 x 40 bf16 = 20 KB
    __shared__ __bf16 Bs[2][BN][LDA];   // 2 x  64 x 40 bf16 = 10 KB

    const int tid  = threadIdx.x;
    const int lane = tid & 31;
    const int wave = tid >> 5;              // 0..7
    const int wmb  = (wave & 3) * 32;       // wave M origin in tile
    const int wnb  = (wave >> 2) * 32;      // wave N origin in tile

    const int m0 = blockIdx.x * BM;
    const int n0 = blockIdx.y * BN;

    const int khalf = lane >> 4;            // K half-group owned by this lane
    const int l16   = lane & 15;

    v8f acc[2][2] = {};                     // 2x2 16x16 f32 tiles

    float4 pa[4], pb[2];
    gload(X, W, m0, n0, K, 0, tid, pa, pb);
    lstore(As[0], Bs[0], tid, pa, pb);

    const int nk = K / BK;
    for (int kk = 0; kk < nk; ++kk) {
        __syncthreads();                    // LDS[cur] ready; prev reads done
        const int  cur  = kk & 1;
        const bool more = (kk + 1) < nk;
        if (more) gload(X, W, m0, n0, K, (kk + 1) * BK, tid, pa, pb);

        // A 16x32 bf16 frag (ISA 7.12.2): lane row m=l16, K = 8*khalf+e / 16+8*khalf+e
        v16bf afrag[2];
#pragma unroll
        for (int t = 0; t < 2; ++t) {
            const int mr = wmb + t * 16 + l16;
#pragma unroll
            for (int e = 0; e < 8; ++e) {
                afrag[t][e]     = As[cur][mr][khalf * 8 + e];
                afrag[t][e + 8] = As[cur][mr][16 + khalf * 8 + e];
            }
        }
        // B 32x16 bf16 frag: n = l16, K = 8*khalf+e / 16+8*khalf+e (same as A)
        v16bf bfrag[2];
#pragma unroll
        for (int u = 0; u < 2; ++u) {
            const int nc = wnb + u * 16 + l16;
#pragma unroll
            for (int e = 0; e < 16; ++e)
                bfrag[u][e] = Bs[cur][nc][(e < 8) ? (khalf * 8 + e) : (16 + khalf * 8 + (e - 8))];
        }

#pragma unroll
        for (int t = 0; t < 2; ++t)
#pragma unroll
            for (int u = 0; u < 2; ++u)
                acc[t][u] = __builtin_amdgcn_wmma_f32_16x16x32_bf16(
                    false, afrag[t], false, bfrag[u],
                    (short)0, acc[t][u], false, false);

        if (more) lstore(As[(kk + 1) & 1], Bs[(kk + 1) & 1], tid, pa, pb);
    }

    // ---- epilogue: folded BatchNorm + bias + ReLU ----
#pragma unroll
    for (int u = 0; u < 2; ++u) {
        const int n = n0 + wnb + u * 16 + l16;
        const float s = gamma[n] * rsqrtf(var[n] + BN_EPS);
        const float t0 = (bias[n] - mean[n]) * s + beta[n];
#pragma unroll
        for (int t = 0; t < 2; ++t) {
            const int mbase = m0 + wmb + t * 16 + (khalf << 3);
#pragma unroll
            for (int r = 0; r < 8; ++r) {
                float y = acc[t][u][r] * s + t0;
                y = y > 0.0f ? y : 0.0f;
                Y[(size_t)(mbase + r) * D_OUT + n] = y;
            }
        }
    }
}

// ---------------------------------------------------------------------------
// Top-3 nearest neighbors. coord_low for one batch (48 KB) staged into LDS
// with CDNA5 async-to-LDS copies (ASYNCcnt), then each thread scans all 4096
// low points for its high point.
// ---------------------------------------------------------------------------
__global__ __launch_bounds__(256)
void knn_top3_kernel(const float* __restrict__ coord_low,   // [B][N_LOW][3]
                     const float* __restrict__ coord_high,  // [B][N_HIGH][3]
                     int* __restrict__ idx_out,             // [B*N_HIGH][3]
                     float* __restrict__ w_out)             // [B*N_HIGH][3]
{
    __shared__ float cl[N_LOW * 3];    // 48 KB; sole LDS object -> offset 0

    const int tid = threadIdx.x;
    const int hp  = blockIdx.x * 256 + tid;
    const int b   = blockIdx.x / (N_HIGH / 256);

    // async global->LDS staging: 3072 x b128 chunks, 12 per thread
    const float* clg = coord_low + (size_t)b * N_LOW * 3;
    for (int c = tid; c < (N_LOW * 3) / 4; c += 256) {
        unsigned ldsoff = (unsigned)(c * 16);        // LDS byte offset of cl[c*4]
        const float* gp = clg + c * 4;
        asm volatile("global_load_async_to_lds_b128 %0, %1, off"
                     :: "v"(ldsoff), "v"(gp)
                     : "memory");
    }
#if __has_builtin(__builtin_amdgcn_s_wait_asynccnt)
    __builtin_amdgcn_s_wait_asynccnt(0);
#else
    asm volatile("s_wait_asynccnt 0" ::: "memory");
#endif
    __syncthreads();

    const float hx = coord_high[(size_t)hp * 3 + 0];
    const float hy = coord_high[(size_t)hp * 3 + 1];
    const float hz = coord_high[(size_t)hp * 3 + 2];

    float d0 = 3.4e38f, d1 = 3.4e38f, d2 = 3.4e38f;
    int   i0 = 0, i1 = 0, i2 = 0;

    for (int j = 0; j < N_LOW; ++j) {
        const float dx = hx - cl[j * 3 + 0];
        const float dy = hy - cl[j * 3 + 1];
        const float dz = hz - cl[j * 3 + 2];
        const float d  = dx * dx + dy * dy + dz * dz;
        if (d < d2) {
            if (d < d1) {
                if (d < d0) { d2 = d1; i2 = i1; d1 = d0; i1 = i0; d0 = d; i0 = j; }
                else        { d2 = d1; i2 = i1; d1 = d;  i1 = j; }
            } else          { d2 = d;  i2 = j; }
        }
    }

    const float r0 = 1.0f / (d0 + 1e-8f);
    const float r1 = 1.0f / (d1 + 1e-8f);
    const float r2 = 1.0f / (d2 + 1e-8f);
    const float inv = 1.0f / (r0 + r1 + r2);

    const int base = hp * 3;
    const int flb  = b * N_LOW;
    idx_out[base + 0] = flb + i0;  w_out[base + 0] = r0 * inv;
    idx_out[base + 1] = flb + i1;  w_out[base + 1] = r1 * inv;
    idx_out[base + 2] = flb + i2;  w_out[base + 2] = r2 * inv;
}

// ---------------------------------------------------------------------------
// Weighted 3-way gather of fl rows, added onto fh (already in out).
// Block handles 4 high points; 64 lanes/point = 64 float4 channels, coalesced.
// ---------------------------------------------------------------------------
__global__ __launch_bounds__(256)
void interp_add_kernel(const float* __restrict__ fl,    // [M_LOW][D_OUT]
                       const int* __restrict__ idx,     // [M_HIGH][3]
                       const float* __restrict__ w,     // [M_HIGH][3]
                       float* __restrict__ out)         // [M_HIGH][D_OUT]
{
    const int tid = threadIdx.x;
    const int i   = blockIdx.x * 4 + (tid >> 6);    // high point
    const int c   = (tid & 63) << 2;                // channel*4
    const int base = i * 3;
    const int j0 = idx[base + 0], j1 = idx[base + 1], j2 = idx[base + 2];
    const float w0 = w[base + 0], w1 = w[base + 1], w2 = w[base + 2];

    const float4 a = *(const float4*)(fl + (size_t)j0 * D_OUT + c);
    const float4 bb = *(const float4*)(fl + (size_t)j1 * D_OUT + c);
    const float4 cc = *(const float4*)(fl + (size_t)j2 * D_OUT + c);
    float4 o = *(float4*)(out + (size_t)i * D_OUT + c);
    o.x += w0 * a.x + w1 * bb.x + w2 * cc.x;
    o.y += w0 * a.y + w1 * bb.y + w2 * cc.y;
    o.z += w0 * a.z + w1 * bb.z + w2 * cc.z;
    o.w += w0 * a.w + w1 * bb.w + w2 * cc.w;
    *(float4*)(out + (size_t)i * D_OUT + c) = o;
}

// ---------------------------------------------------------------------------
// Passthrough: second tuple output = coord_high
// ---------------------------------------------------------------------------
__global__ __launch_bounds__(256)
void copy_coords_kernel(const float* __restrict__ src, float* __restrict__ dst, int n)
{
    const int i = blockIdx.x * 256 + threadIdx.x;
    if (i < n) dst[i] = src[i];
}

// ---------------------------------------------------------------------------
extern "C" void kernel_launch(void* const* d_in, const int* in_sizes, int n_in,
                              void* d_out, int out_size, void* d_ws, size_t ws_size,
                              hipStream_t stream)
{
    const float* feature_low  = (const float*)d_in[0];
    const float* coord_low    = (const float*)d_in[1];
    const float* feature_high = (const float*)d_in[2];
    const float* coord_high   = (const float*)d_in[3];
    const float* W_low        = (const float*)d_in[4];
    const float* b_low        = (const float*)d_in[5];
    const float* gamma_low    = (const float*)d_in[6];
    const float* beta_low     = (const float*)d_in[7];
    const float* mean_low     = (const float*)d_in[8];
    const float* var_low      = (const float*)d_in[9];
    const float* W_high       = (const float*)d_in[10];
    const float* b_high       = (const float*)d_in[11];
    const float* gamma_high   = (const float*)d_in[12];
    const float* beta_high    = (const float*)d_in[13];
    const float* mean_high    = (const float*)d_in[14];
    const float* var_high     = (const float*)d_in[15];

    float* out_feat   = (float*)d_out;                               // [M_HIGH][D_OUT]
    float* out_coords = (float*)d_out + (size_t)M_HIGH * D_OUT;      // [M_HIGH][3]

    float* fl     = (float*)d_ws;                                    // [M_LOW][D_OUT]
    int*   knn_i  = (int*)((float*)d_ws + FL_ELEMS);                 // [M_HIGH][3]
    float* knn_w  = (float*)d_ws + FL_ELEMS + KNN_ELEMS;             // [M_HIGH][3]

    // 1) fl = relu(BN(feature_low @ W_low^T + b_low))  -> workspace
    {
        dim3 grid(M_LOW / BM, D_OUT / BN);
        mlp_gemm_kernel<<<grid, 256, 0, stream>>>(
            feature_low, W_low, b_low, gamma_low, beta_low, mean_low, var_low,
            fl, M_LOW, K_LOW);
    }
    // 2) fh = relu(BN(feature_high @ W_high^T + b_high)) -> directly into out
    {
        dim3 grid(M_HIGH / BM, D_OUT / BN);
        mlp_gemm_kernel<<<grid, 256, 0, stream>>>(
            feature_high, W_high, b_high, gamma_high, beta_high, mean_high, var_high,
            out_feat, M_HIGH, K_HIGH);
    }
    // 3) second output: coord_high passthrough
    {
        const int n = M_HIGH * 3;
        copy_coords_kernel<<<(n + 255) / 256, 256, 0, stream>>>(coord_high, out_coords, n);
    }
    // 4) top-3 NN + inverse-distance weights (async-LDS staged scan)
    {
        knn_top3_kernel<<<M_HIGH / 256, 256, 0, stream>>>(
            coord_low, coord_high, knn_i, knn_w);
    }
    // 5) out += weighted gather of fl
    {
        interp_add_kernel<<<M_HIGH / 4, 256, 0, stream>>>(fl, knn_i, knn_w, out_feat);
    }
}
